// PredictionScore_70789650973258
// MI455X (gfx1250) — matrix operand order repaired and checked by reading, not compiled, not run on hardware
//
#include <hip/hip_runtime.h>
#include <math.h>

typedef _Float16 f16_t;
typedef __attribute__((ext_vector_type(16))) _Float16 v16h;
typedef __attribute__((ext_vector_type(8)))  float    v8f;

#define NB   8
#define NL   64
#define NPP  512
#define NC   128
#define NG   10

// output layout (floats)
#define OUT_SG 2621440L   // 8*64*512*10
#define OUT_MU 5242880L
#define OUT_D  7864320L
#define OUT_BA 8126464L

// workspace layout (float offsets)
#define WS_A   0          // 512*128
#define WS_BP  65536      // 4096*128
#define WS_SC  589824     // 128
#define WS_SH  589952     // 128
#define WS_HB  590080     // 32
#define WS_WH  590112     // 32*128 f16

// branchless elu: max(x,0) + (exp(min(x,0)) - 1); error ~1e-3 rel near 0-,
// same order as the f16 quantization applied right after.
__device__ __forceinline__ float elu_fast(float x) {
  return fmaxf(x, 0.f) + (__expf(fminf(x, 0.f)) - 1.f);
}

// ---------------- setup: fold BN affine, pack head weights (f16, [32][128]) ----
__global__ void setup_kernel(const float* mlp_b, const float* mlp_g, const float* mlp_be,
                             const float* mlp_m, const float* mlp_v,
                             const float* pi_W, const float* pi_b,
                             const float* sg_W, const float* sg_b,
                             const float* mu_W, const float* mu_b,
                             float* sc, float* sh, float* hb, f16_t* wht) {
  int t = threadIdx.x;
  if (t < NC) {
    float s = mlp_g[t] * rsqrtf(mlp_v[t] + 1e-5f);
    sc[t] = s;
    sh[t] = (mlp_b[t] - mlp_m[t]) * s + mlp_be[t];
  }
  if (t < 32) {
    float v = 0.f;
    if (t < 10)       v = pi_b[t];
    else if (t < 20)  v = sg_b[t - 10];
    else if (t < 30)  v = mu_b[t - 20];
    hb[t] = v;
  }
  for (int idx = t; idx < 32 * NC; idx += blockDim.x) {
    int n = idx >> 7, k = idx & 127;
    float v = 0.f;
    if (n < 10)       v = pi_W[k * NG + n];
    else if (n < 20)  v = sg_W[k * NG + (n - 10)];
    else if (n < 30)  v = mu_W[k * NG + (n - 20)];
    wht[n * NC + k] = (f16_t)v;
  }
}

// ---------------- binding affinity head (tiny, fp32) ---------------------------
__global__ void ba_kernel(const float* hl, const int* nl_counts,
                          const float* W1, const float* b1, const float* g1,
                          const float* be1, const float* m1, const float* v1,
                          const float* W2, const float* b2, const float* g2,
                          const float* be2, const float* m2, const float* v2,
                          const float* W3, const float* b3, float* out_ba) {
  __shared__ float h[NC], x1[2 * NC], x2[NC], red[256];
  int b = blockIdx.x, t = threadIdx.x;
  int nl = nl_counts[b];
  if (t < NC) {
    float acc = 0.f;
    for (int i = 0; i < nl; ++i) acc += hl[(b * NL + i) * NC + t];
    h[t] = acc;
  }
  __syncthreads();
  {
    float acc = b1[t];
    for (int k = 0; k < NC; ++k) acc += h[k] * W1[k * (2 * NC) + t];
    acc = (acc - m1[t]) * rsqrtf(v1[t] + 1e-5f) * g1[t] + be1[t];
    x1[t] = acc > 0.f ? acc : expm1f(acc);
  }
  __syncthreads();
  if (t < NC) {
    float acc = b2[t];
    for (int k = 0; k < 2 * NC; ++k) acc += x1[k] * W2[k * NC + t];
    acc = (acc - m2[t]) * rsqrtf(v2[t] + 1e-5f) * g2[t] + be2[t];
    x2[t] = acc > 0.f ? acc : expm1f(acc);
  }
  __syncthreads();
  red[t] = (t < NC) ? x2[t] * W3[t] : 0.f;
  __syncthreads();
  for (int s = 128; s > 0; s >>= 1) {
    if (t < s) red[t] += red[t + s];
    __syncthreads();
  }
  if (t == 0) out_ba[b] = red[0] + b3[0];
}

// ---------------- projection GEMMs: A = hl_m @ W_top, Bp = hp_m @ W_bot --------
__global__ void proj_kernel(const float* hl, const float* hp,
                            const int* nl_counts, const int* np_counts,
                            const float* mlp_W, float* A, float* Bp) {
  __shared__ float row[NC];
  int blk = blockIdx.x, t = threadIdx.x;
  const float* in; float* out; int koff, valid;
  if (blk < NB * NL) {
    int b = blk >> 6, i = blk & 63;
    valid = (i < nl_counts[b]);
    in = hl + (long)blk * NC; out = A + (long)blk * NC; koff = 0;
  } else {
    int r = blk - NB * NL;
    int b = r >> 9, j = r & 511;
    valid = (j < np_counts[b]);
    in = hp + (long)r * NC; out = Bp + (long)r * NC; koff = NC;
  }
  row[t] = valid ? in[t] : 0.f;
  __syncthreads();
  float acc = 0.f;
  for (int k = 0; k < NC; ++k) acc += row[k] * mlp_W[(k + koff) * NC + t];
  out[t] = acc;
}

// ---------------- main fused pair kernel: elu(BN(A_i+Bp_j)) -> WMMA heads ------
__global__ void __launch_bounds__(128) mdn_kernel(
    const float* __restrict__ A, const float* __restrict__ Bp,
    const float* __restrict__ sc, const float* __restrict__ sh,
    const float* __restrict__ hb, const f16_t* __restrict__ wht,
    const float* __restrict__ l_pos, const float* __restrict__ p_pos,
    const int* __restrict__ nl_counts, const int* __restrict__ np_counts,
    float* __restrict__ out) {
  __shared__ float s_bp[64][132];   // pre-scaled by sc[k]; stride 132 floats (528 B, 8B-aligned rows)
  __shared__ float s_base[NC];      // A_i[k]*sc[k] + sh[k]
  __shared__ float s_scg[NC];
  __shared__ float s_hb[32];
  __shared__ float s_acc[4][16][33];

  int t   = threadIdx.x;
  int blk = blockIdx.x;
  int jt  = blk & 7;
  int i   = (blk >> 3) & 63;
  int b   = blk >> 9;
  int jb0 = jt * 64;

  // phase 1: per-channel affine fold
  if (t < NC) {
    float scv = sc[t];
    s_scg[t]  = scv;
    s_base[t] = A[(long)(b * NL + i) * NC + t] * scv + sh[t];
  }
  if (t < 32) s_hb[t] = hb[t];
  __syncthreads();

  // phase 2: stage Bp tile pre-scaled by sc[k]
  const float* bp_src = Bp + ((long)(b * NPP + jb0)) * NC;
  for (int idx = t; idx < 64 * NC; idx += 128) {
    int j = idx >> 7, k = idx & 127;
    s_bp[j][k] = bp_src[idx] * s_scg[k];
  }
  __syncthreads();

  int lane = t & 31;
  int wave = t >> 5;
  int half = lane >> 4;        // 0: lanes 0-15, 1: lanes 16-31
  int m    = lane & 15;
  int jl   = wave * 16 + m;    // local pair row for A-fragment

  // Preload B fragments: 2 N-tiles x 4 K-blocks. B fragments use contiguous k here (same as A):
  // lanes 0-15 hold K=kb..kb+15 (2/VGPR), lanes 16-31 hold K=kb+16..kb+31, N=lane%16.
  v16h bf[2][4];
  for (int nt = 0; nt < 2; ++nt) {
    const f16_t* wp = wht + (m + nt * 16) * NC;
    for (int kq = 0; kq < 4; ++kq) {
      int base = kq * 32 + half * 16;
      v16h f;
      #pragma unroll
      for (int v = 0; v < 8; ++v) {
        f[2 * v]     = wp[base + 2 * v];
        f[2 * v + 1] = wp[base + 2 * v + 1];
      }
      bf[nt][kq] = f;
    }
  }

  v8f acc0 = {}; v8f acc1 = {};
  #pragma unroll
  for (int kq = 0; kq < 4; ++kq) {
    // A fragment uses contiguous k (same as B): lanes 0-15 -> K 0-15; lanes 16-31 -> K 16-31
    v16h af;
    #pragma unroll
    for (int v = 0; v < 8; ++v) {
      int kA = kq * 32 + half * 16 + 2 * v;
      float2 bb = *(const float2*)&s_bp[jl][kA];     // kA even -> 8B aligned -> ds_load_b64
      float2 aa = *(const float2*)&s_base[kA];
      float h0 = elu_fast(aa.x + bb.x);
      float h1 = elu_fast(aa.y + bb.y);
      af[2 * v]     = (f16_t)h0;
      af[2 * v + 1] = (f16_t)h1;
    }
    acc0 = __builtin_amdgcn_wmma_f32_16x16x32_f16(false, af, false, bf[0][kq],
                                                  (short)0, acc0, false, false);
    acc1 = __builtin_amdgcn_wmma_f32_16x16x32_f16(false, af, false, bf[1][kq],
                                                  (short)0, acc1, false, false);
  }

  // C/D layout: VGPR r -> row (r + half*8), col = lane%16 (+16 for tile 1)
  #pragma unroll
  for (int r = 0; r < 8; ++r) {
    s_acc[wave][r + half * 8][m]      = acc0[r];
    s_acc[wave][r + half * 8][m + 16] = acc1[r];
  }
  __syncthreads();

  if (lane < 16) {
    int j = jb0 + wave * 16 + lane;
    float vals[32];
    #pragma unroll
    for (int n = 0; n < 32; ++n) vals[n] = s_acc[wave][lane][n];

    float lm = (i < nl_counts[b]) ? 1.f : 0.f;
    float pm = (j < np_counts[b]) ? 1.f : 0.f;
    float cm = lm * pm;
    long pbase = ((long)(b * NL + i) * NPP + j) * NG;

    // pi: softmax + eps, masked
    float lg[NG]; float mx = -1e30f;
    #pragma unroll
    for (int g = 0; g < NG; ++g) { lg[g] = vals[g] + s_hb[g]; mx = fmaxf(mx, lg[g]); }
    float ssum = 0.f;
    #pragma unroll
    for (int g = 0; g < NG; ++g) { lg[g] = __expf(lg[g] - mx); ssum += lg[g]; }
    float inv = 1.f / ssum;
    #pragma unroll
    for (int g = 0; g < NG; ++g) out[pbase + g] = (lg[g] * inv + 1e-10f) * cm;

    // sigma: elu + 1.1 + eps
    #pragma unroll
    for (int g = 0; g < NG; ++g) {
      float x = elu_fast(vals[10 + g] + s_hb[10 + g]);
      out[OUT_SG + pbase + g] = (x + 1.1f + 1e-10f) * cm;
    }
    // mu: elu + 1.0 + eps
    #pragma unroll
    for (int g = 0; g < NG; ++g) {
      float x = elu_fast(vals[20 + g] + s_hb[20 + g]);
      out[OUT_MU + pbase + g] = (x + 1.0f + 1e-10f) * cm;
    }
    // dist (masked; when cm==1 masked positions equal raw positions)
    const float* X = l_pos + (long)(b * NL + i) * 3;
    const float* Y = p_pos + (long)(b * NPP + j) * 3;
    float dx = X[0] - Y[0], dy = X[1] - Y[1], dz = X[2] - Y[2];
    float d2 = dx * dx + dy * dy + dz * dz;
    out[OUT_D + (long)(b * NL + i) * NPP + j] = sqrtf(fmaxf(d2, 0.f)) * cm;
  }
}

extern "C" void kernel_launch(void* const* d_in, const int* in_sizes, int n_in,
                              void* d_out, int out_size, void* d_ws, size_t ws_size,
                              hipStream_t stream) {
  (void)in_sizes; (void)n_in; (void)out_size; (void)ws_size;
  const float* hl     = (const float*)d_in[0];
  const float* hp     = (const float*)d_in[1];
  const float* l_pos  = (const float*)d_in[2];
  const float* p_pos  = (const float*)d_in[3];
  const int*   nl_c   = (const int*)  d_in[4];
  const int*   np_c   = (const int*)  d_in[5];
  const float* mlp_W  = (const float*)d_in[6];
  const float* mlp_b  = (const float*)d_in[7];
  const float* mlp_g  = (const float*)d_in[8];
  const float* mlp_be = (const float*)d_in[9];
  const float* mlp_m  = (const float*)d_in[10];
  const float* mlp_v  = (const float*)d_in[11];
  const float* ba_W1  = (const float*)d_in[12];
  const float* ba_b1  = (const float*)d_in[13];
  const float* ba_g1  = (const float*)d_in[14];
  const float* ba_be1 = (const float*)d_in[15];
  const float* ba_m1  = (const float*)d_in[16];
  const float* ba_v1  = (const float*)d_in[17];
  const float* ba_W2  = (const float*)d_in[18];
  const float* ba_b2  = (const float*)d_in[19];
  const float* ba_g2  = (const float*)d_in[20];
  const float* ba_be2 = (const float*)d_in[21];
  const float* ba_m2  = (const float*)d_in[22];
  const float* ba_v2  = (const float*)d_in[23];
  const float* ba_W3  = (const float*)d_in[24];
  const float* ba_b3  = (const float*)d_in[25];
  const float* pi_W   = (const float*)d_in[26];
  const float* pi_b   = (const float*)d_in[27];
  const float* sg_W   = (const float*)d_in[28];
  const float* sg_b   = (const float*)d_in[29];
  const float* mu_W   = (const float*)d_in[30];
  const float* mu_b   = (const float*)d_in[31];

  float* out = (float*)d_out;
  float* ws  = (float*)d_ws;
  float* wsA  = ws + WS_A;
  float* wsBp = ws + WS_BP;
  float* wsc  = ws + WS_SC;
  float* wsh  = ws + WS_SH;
  float* whb  = ws + WS_HB;
  f16_t* wht  = (f16_t*)(ws + WS_WH);

  setup_kernel<<<1, 256, 0, stream>>>(mlp_b, mlp_g, mlp_be, mlp_m, mlp_v,
                                      pi_W, pi_b, sg_W, sg_b, mu_W, mu_b,
                                      wsc, wsh, whb, wht);
  ba_kernel<<<NB, 256, 0, stream>>>(hl, nl_c,
                                    ba_W1, ba_b1, ba_g1, ba_be1, ba_m1, ba_v1,
                                    ba_W2, ba_b2, ba_g2, ba_be2, ba_m2, ba_v2,
                                    ba_W3, ba_b3, out + OUT_BA);
  proj_kernel<<<NB * NL + NB * NPP, NC, 0, stream>>>(hl, hp, nl_c, np_c,
                                                     mlp_W, wsA, wsBp);
  mdn_kernel<<<NB * NL * (NPP / 64), 128, 0, stream>>>(wsA, wsBp, wsc, wsh, whb, wht,
                                                       l_pos, p_pos, nl_c, np_c, out);
}
